// ConductanceLIFNetwork_IO_90013924590461
// MI455X (gfx1250) — compile-verified, matrix operand order checked by reading, not run on hardware
//
#include <hip/hip_runtime.h>
#include <hip/hip_bf16.h>
#include <stdint.h>

typedef __attribute__((ext_vector_type(16))) __bf16       v16bf_t;
typedef __attribute__((ext_vector_type(8)))  float        v8f_t;

#define N_NEUR  4096
#define N_IN    1024
#define N_B     8
#define N_T     500
#define N_TILES 256      /* N_NEUR/16 */
#define RQ      128      /* recurrent K chunks of 32 */
#define FQ      32       /* feedforward K chunks of 32 */
#define STAGE   16       /* K chunks staged in LDS per barrier round */
#define INST    (N_NEUR*N_B)   /* 32768 */

/* double-exponential kinetics constants: exp(-DT/tau) */
#define AR0 0.81873075307798182f   /* exp(-0.2)   tau_rise 0.5  */
#define AR1 0.95122942450071402f   /* exp(-0.05)  tau_rise 2.0  */
#define AR2 0.81873075307798182f
#define AR3 0.81873075307798182f
#define AD0 0.95122942450071402f   /* exp(-0.05)  tau_dec 2.0   */
#define AD1 0.99900049983337499f   /* exp(-0.001) tau_dec 100   */
#define AD2 0.98019867330675525f   /* exp(-0.02)  tau_dec 5.0   */
#define AD3 0.95122942450071402f

static __device__ __forceinline__ unsigned short f2bf(float f) {
  unsigned u = __float_as_uint(f);
  unsigned r = (u + 0x7FFFu + ((u >> 16) & 1u)) >> 16;
  return (unsigned short)r;
}

/* expand 2 spike bits at WMMA K-position into a packed bf16 {0,1} pair */
static __device__ __forceinline__ void build_a_words(unsigned x, unsigned hi,
                                                     unsigned* __restrict__ dst) {
#pragma unroll
  for (int r = 0; r < 8; ++r) {
    unsigned K0 = ((2u * r) & 7u) + ((unsigned)(r >> 2) << 4) + (hi << 3);
    unsigned k2 = (x >> K0) & 3u;
    dst[r] = ((0u - (k2 & 1u)) & 0x3F80u) | ((0u - (k2 >> 1)) & 0x3F800000u);
  }
}

/* ---- prep: swizzle recurrent weights (f32 -> bf16, WMMA-B per-lane layout) ----
 * element index: idx = ((tile*128 + q)*32 + lane)*16 + e
 * holds W_eff[k = q*32 + K(e,lane)][j = tile*16 + (lane&15)]                      */
__global__ __launch_bounds__(256) void pack_w_kernel(
    const float* __restrict__ W, const float* __restrict__ scal,
    const int* __restrict__ ci, unsigned short* __restrict__ dst) {
  unsigned idx = blockIdx.x * 256u + threadIdx.x;           /* 16777216 total */
  unsigned e    = idx & 15u;
  unsigned lane = (idx >> 4) & 31u;
  unsigned q    = (idx >> 9) & 127u;
  unsigned tile = idx >> 16;
  unsigned j = tile * 16u + (lane & 15u);
  unsigned K = (e & 7u) + ((e >> 3) << 4) + ((lane >> 4) << 3);
  unsigned k = q * 32u + K;
  float w  = W[(size_t)k * N_NEUR + j];
  float sc = scal[ci[k] * 2 + ci[j]];
  dst[idx] = f2bf(w * sc);
}

/* ---- prep: swizzle feedforward weights ---- */
__global__ __launch_bounds__(256) void pack_wff_kernel(
    const float* __restrict__ Wff, const float* __restrict__ scff,
    const int* __restrict__ ci, unsigned short* __restrict__ dst) {
  unsigned idx = blockIdx.x * 256u + threadIdx.x;           /* 4194304 total */
  unsigned e    = idx & 15u;
  unsigned lane = (idx >> 4) & 31u;
  unsigned q    = (idx >> 9) & 31u;
  unsigned tile = idx >> 14;
  unsigned j = tile * 16u + (lane & 15u);
  unsigned K = (e & 7u) + ((e >> 3) << 4) + ((lane >> 4) << 3);
  unsigned k = q * 32u + K;
  float w  = Wff[(size_t)k * N_NEUR + j];
  float sc = scff[ci[j]];                                   /* ci_ff all zero */
  dst[idx] = f2bf(w * sc);
}

/* ---- prep: bit-pack input spikes: word(b,t,q) bit i = input[b,t,q*32+i] ---- */
__global__ __launch_bounds__(256) void pack_inbits_kernel(
    const float* __restrict__ in, unsigned* __restrict__ dst) {
  unsigned idx = blockIdx.x * 256u + threadIdx.x;
  if (idx >= (unsigned)(N_B * N_T * FQ)) return;            /* 128000 */
  unsigned q = idx & 31u;
  unsigned t = (idx >> 5) % N_T;
  unsigned b = idx / (N_T * FQ);
  const float* p = in + ((size_t)b * N_T + t) * N_IN + q * 32u;
  unsigned bits = 0;
#pragma unroll 4
  for (int i = 0; i < 32; ++i) bits |= (p[i] > 0.5f) ? (1u << i) : 0u;
  dst[idx] = bits;
}

/* ---- prep: ci bitmask + state init + spike-bit clear ---- */
__global__ __launch_bounds__(256) void init_state_kernel(
    const int* __restrict__ ci, unsigned* __restrict__ cibits,
    float* __restrict__ v, float* __restrict__ ref,
    float* __restrict__ h, float* __restrict__ g,
    unsigned short* __restrict__ sb) {
  unsigned idx = blockIdx.x * 256u + threadIdx.x;
  if (idx < INST) {
    v[idx] = -70.0f;   /* E_L for both cell types */
    ref[idx] = 0.0f;
#pragma unroll
    for (int s = 0; s < 4; ++s) { h[s * INST + idx] = 0.0f; g[s * INST + idx] = 0.0f; }
  }
  if (idx < RQ) {
    unsigned bits = 0;
#pragma unroll 4
    for (int i = 0; i < 32; ++i) bits |= (ci[idx * 32 + i] == 1) ? (1u << i) : 0u;
    cibits[idx] = bits;
  }
  if (idx < 2u * N_B * N_TILES) sb[idx] = 0;                /* both ping-pong buffers */
}

/* ---- one timestep: staged LDS A-operand + WMMA drive + LIF state update ----
 * grid 32 x 256 threads = 8 waves/block, one 16-neuron tile per wave.
 * A tile (spikes) is identical for all waves -> built cooperatively in LDS.
 * Accumulator M packing: M = cell_type*8 + batch (exc rows 0-7, inh rows 8-15). */
__global__ __launch_bounds__(256) void snn_step_kernel(
    int t,
    const uint4* __restrict__ Wswz, const uint4* __restrict__ Wffswz,
    const unsigned* __restrict__ inbits, const unsigned* __restrict__ cibits,
    const int* __restrict__ ci,
    float* __restrict__ vbuf, float* __restrict__ refbuf,
    float* __restrict__ hbuf, float* __restrict__ gbuf,
    const unsigned short* __restrict__ sbin, unsigned short* __restrict__ sbout,
    float* __restrict__ out) {
  __shared__ unsigned a_lds[STAGE * 32 * 8];   /* 16 KB: STAGE chunks x 32 lanes x 32B */

  const int tid  = threadIdx.x;
  const int lane = tid & 31;
  const int tile = blockIdx.x * 8 + (tid >> 5);
  const int m  = lane & 15;
  const int hi = lane >> 4;
  const int b_lane = m & 7;

  v8f_t cacc = {0, 0, 0, 0, 0, 0, 0, 0};
  v8f_t facc = {0, 0, 0, 0, 0, 0, 0, 0};

  union AU { uint4 q2[2]; v16bf_t m; };

  /* ---- recurrent drive: D[16x16] += A(spikes,16x32) x B(W,32x16), K = 4096 ---- */
  {
    const unsigned* sw_base = (const unsigned*)sbin;        /* 128 u32 per batch */
    const uint4* wb = Wswz + ((size_t)tile * RQ * 32 + lane) * 2;
    for (int s0 = 0; s0 < RQ; s0 += STAGE) {
      /* cooperative build: 256 threads x 2 = STAGE*32 (chunk,lane) cells */
#pragma unroll
      for (int s = 0; s < (STAGE * 32) / 256; ++s) {
        int cell = s * 256 + tid;
        int ql = cell >> 5;          /* chunk within stage   */
        int bl = cell & 31;          /* builder's lane role  */
        int bm = bl & 15, bhi = bl >> 4;
        int bb = bm & 7, bc = bm >> 3;
        int q = s0 + ql;
        unsigned sw = sw_base[bb * 128 + q];
        unsigned cw = cibits[q];
        unsigned x = sw & (bc ? cw : ~cw);   /* mask spikes by presyn cell type */
        build_a_words(x, (unsigned)bhi, &a_lds[(size_t)cell * 8]);
      }
      __syncthreads();
      /* consume: each wave runs its tile's weight slab over the staged A */
#pragma unroll 4
      for (int ql = 0; ql < STAGE; ++ql) {
        const uint4* ap = (const uint4*)&a_lds[((size_t)ql * 32 + lane) * 8];
        AU a; a.q2[0] = ap[0]; a.q2[1] = ap[1];
        AU b; b.q2[0] = wb[0]; b.q2[1] = wb[1]; wb += 64;
        cacc = __builtin_amdgcn_wmma_f32_16x16x32_bf16(false, a.m, false, b.m,
                                                       (short)0, cacc, false, false);
      }
      __syncthreads();
    }
  }

  /* ---- feedforward drive: rows 0-7 = batch, rows 8-15 zero, K = 1024 ---- */
  {
    const uint4* wb = Wffswz + ((size_t)tile * FQ * 32 + lane) * 2;
    for (int s0 = 0; s0 < FQ; s0 += STAGE) {
#pragma unroll
      for (int s = 0; s < (STAGE * 32) / 256; ++s) {
        int cell = s * 256 + tid;
        int ql = cell >> 5;
        int bl = cell & 31;
        int bm = bl & 15, bhi = bl >> 4;
        int bb = bm & 7;
        int q = s0 + ql;
        unsigned x = inbits[((size_t)bb * N_T + t) * FQ + q];
        x &= (bm < 8) ? 0xFFFFFFFFu : 0u;    /* zero rows 8-15 */
        build_a_words(x, (unsigned)bhi, &a_lds[(size_t)cell * 8]);
      }
      __syncthreads();
#pragma unroll 4
      for (int ql = 0; ql < STAGE; ++ql) {
        const uint4* ap = (const uint4*)&a_lds[((size_t)ql * 32 + lane) * 8];
        AU a; a.q2[0] = ap[0]; a.q2[1] = ap[1];
        AU b; b.q2[0] = wb[0]; b.q2[1] = wb[1]; wb += 64;
        facc = __builtin_amdgcn_wmma_f32_16x16x32_bf16(false, a.m, false, b.m,
                                                       (short)0, facc, false, false);
      }
      __syncthreads();
    }
  }

  /* ---- cross-half exchange: exc rows live in lanes 0-15, inh in 16-31 ---- */
  float sx[8], fx[4];
#pragma unroll
  for (int r = 0; r < 8; ++r) sx[r] = __shfl_xor(cacc[r], 16, 32);
#pragma unroll
  for (int r = 0; r < 4; ++r) fx[r] = __shfl_xor(facc[r + 4], 16, 32);

  /* lane handles neuron n = tile*16 + m, batches b = hi*4 + i, i=0..3 */
  const int n = tile * 16 + m;
  const int ct = ci[n];
  const float dt_tau   = ct ? 0.01f : 0.005f;   /* DT/tau_mem */
  const float refsteps = ct ? 10.0f : 20.0f;
  const float theta = -50.0f, u_reset = -65.0f, e_l = -70.0f, inv_gl = 0.1f;

  const size_t base = (size_t)n * 8 + (size_t)hi * 4;
  float4 v4 = *(const float4*)(vbuf + base);
  float4 r4 = *(const float4*)(refbuf + base);
  float4 h4[4], g4[4];
#pragma unroll
  for (int s = 0; s < 4; ++s) {
    h4[s] = *(const float4*)(hbuf + (size_t)s * INST + base);
    g4[s] = *(const float4*)(gbuf + (size_t)s * INST + base);
  }
  float* vp = &v4.x; float* rp = &r4.x;
  int sp[4];

#pragma unroll
  for (int i = 0; i < 4; ++i) {
    float exc = hi ? sx[4 + i] : cacc[i];
    float inh = hi ? cacc[4 + i] : sx[i];
    float dff = hi ? fx[i] : facc[i];
    float* hp0 = (&h4[0].x) + i; float* hp1 = (&h4[1].x) + i;
    float* hp2 = (&h4[2].x) + i; float* hp3 = (&h4[3].x) + i;
    float* gp0 = (&g4[0].x) + i; float* gp1 = (&g4[1].x) + i;
    float* gp2 = (&g4[2].x) + i; float* gp3 = (&g4[3].x) + i;
    /* drive per channel: 0: exc*G_BAR0(1.0), 1: exc*G_BAR1(0.5), 2: inh, 3: ff */
    float h0 = *hp0 * AR0 + exc;
    float h1 = *hp1 * AR1 + 0.5f * exc;
    float h2 = *hp2 * AR2 + inh;
    float h3 = *hp3 * AR3 + dff;
    float g0 = *gp0 * AD0 + (1.0f - AD0) * h0;
    float g1 = *gp1 * AD1 + (1.0f - AD1) * h1;
    float g2 = *gp2 * AD2 + (1.0f - AD2) * h2;
    float g3 = *gp3 * AD3 + (1.0f - AD3) * h3;
    float vv = vp[i];
    float I = g0 * (0.0f - vv) + g1 * (0.0f - vv) + g2 * (-80.0f - vv) + g3 * (0.0f - vv);
    float vn = vv + dt_tau * ((e_l - vv) + I * inv_gl);
    float rr = rp[i];
    bool refr = rr > 0.0f;
    vn = refr ? u_reset : vn;
    bool spk = (vn - theta > 0.0f) && !refr;
    float snew = spk ? 1.0f : 0.0f;
    vp[i] = spk ? u_reset : vn;
    rp[i] = spk ? refsteps : fmaxf(rr - 1.0f, 0.0f);
    *hp0 = h0; *hp1 = h1; *hp2 = h2; *hp3 = h3;
    *gp0 = g0; *gp1 = g1; *gp2 = g2; *gp3 = g3;
    sp[i] = spk ? 1 : 0;
    int b = hi * 4 + i;
    out[((size_t)b * N_T + t) * N_NEUR + n] = snew;
  }

  *(float4*)(vbuf + base) = v4;
  *(float4*)(refbuf + base) = r4;
#pragma unroll
  for (int s = 0; s < 4; ++s) {
    *(float4*)(hbuf + (size_t)s * INST + base) = h4[s];
    *(float4*)(gbuf + (size_t)s * INST + base) = g4[s];
  }

  /* bit-pack this tile's spikes for next step: ballot bit L<16 -> b=i, L>=16 -> b=i+4 */
#pragma unroll
  for (int i = 0; i < 4; ++i) {
    unsigned mask = (unsigned)__ballot(sp[i]);
    if (lane == 0)  sbout[(size_t)i * N_TILES + tile]       = (unsigned short)(mask & 0xFFFFu);
    if (lane == 16) sbout[(size_t)(i + 4) * N_TILES + tile] = (unsigned short)(mask >> 16);
  }
}

/* ---------------- workspace layout (bytes) ---------------- */
#define WS_WSWZ   0u                             /* 33554432  bf16 swizzled W   */
#define WS_WFF    33554432u                      /* 8388608   bf16 swizzled Wff */
#define WS_INB    41943040u                      /* 512000    input spike bits  */
#define WS_CIB    42455296u                      /* 512       ci bitmask        */
#define WS_V      42455808u                      /* 131072                      */
#define WS_REF    42586880u                      /* 131072                      */
#define WS_H      42717952u                      /* 524288                      */
#define WS_G      43242240u                      /* 524288                      */
#define WS_SB     43766528u                      /* 8192 (2 x 4096B ping-pong)  */

extern "C" void kernel_launch(void* const* d_in, const int* in_sizes, int n_in,
                              void* d_out, int out_size, void* d_ws, size_t ws_size,
                              hipStream_t stream) {
  const float* in_spk = (const float*)d_in[0];   /* (8,500,1024) */
  const float* W      = (const float*)d_in[1];   /* (4096,4096)  */
  const float* Wff    = (const float*)d_in[2];   /* (1024,4096)  */
  const float* scal   = (const float*)d_in[3];   /* (2,2)        */
  const float* scff   = (const float*)d_in[4];   /* (1,2)        */
  const int*   ci     = (const int*)d_in[5];     /* (4096,)      */
  /* d_in[6] = ci_ff, all zeros, folded into pack_wff */

  char* ws = (char*)d_ws;
  unsigned short* wswz = (unsigned short*)(ws + WS_WSWZ);
  unsigned short* wff  = (unsigned short*)(ws + WS_WFF);
  unsigned*       inb  = (unsigned*)(ws + WS_INB);
  unsigned*       cib  = (unsigned*)(ws + WS_CIB);
  float* v   = (float*)(ws + WS_V);
  float* ref = (float*)(ws + WS_REF);
  float* h   = (float*)(ws + WS_H);
  float* g   = (float*)(ws + WS_G);
  unsigned short* sb = (unsigned short*)(ws + WS_SB);
  float* out = (float*)d_out;

  pack_w_kernel   <<<16777216 / 256, 256, 0, stream>>>(W, scal, ci, wswz);
  pack_wff_kernel <<<4194304 / 256, 256, 0, stream>>>(Wff, scff, ci, wff);
  pack_inbits_kernel<<<(N_B * N_T * FQ + 255) / 256, 256, 0, stream>>>(in_spk, inb);
  init_state_kernel <<<(INST + 255) / 256, 256, 0, stream>>>(ci, cib, v, ref, h, g, sb);

  const unsigned sb_words = N_B * N_TILES;   /* 2048 u16 per buffer */
  for (int t = 0; t < N_T; ++t) {
    const unsigned short* sbin = sb + (size_t)(t & 1) * sb_words;
    unsigned short* sbout      = sb + (size_t)((t + 1) & 1) * sb_words;
    snn_step_kernel<<<32, 256, 0, stream>>>(
        t, (const uint4*)wswz, (const uint4*)wff, inb, cib, ci,
        v, ref, h, g, sbin, sbout, out);
  }
}
